// RANCoders_29695403885050
// MI455X (gfx1250) — matrix operand order repaired and checked by reading, not compiled, not run on hardware
//
#include <hip/hip_runtime.h>
#include <stdint.h>

// RANCoders ensemble forward on gfx1250.
// Every dense layer is one (or more) v_wmma_f32_16x16x32_f16, M=16-row tiles,
// K zero-padded to 32, N padded to multiples of 16. Weights are transposed and
// zero-padded into the exact WMMA B-matrix LDS layout once per block; per-wave
// activation tiles live in LDS in the WMMA A-matrix-friendly row-major [16][32]
// f16 layout. Output-bandwidth bound (419 MB of fp32 stores), so compute is
// kept entirely on the matrix pipe.

typedef _Float16 v8h  __attribute__((ext_vector_type(8)));
typedef _Float16 v16h __attribute__((ext_vector_type(16)));
typedef float    v8f  __attribute__((ext_vector_type(8)));

namespace {

constexpr int kE  = 100;   // estimators
constexpr int kMF = 16;    // max_features
constexpr int kF  = 64;    // input features
constexpr int kB  = 8192;  // batch
constexpr int kL  = 8;     // latent

constexpr int NTHREADS       = 256;  // 8 waves (wave32)
constexpr int ROWS_PER_BLOCK = 128;  // 8 waves * 16 rows

// f16 weight LDS offsets; each matrix stored transposed [npad][32] (K padded to 32)
constexpr int W_ENC0 = 0;                 // 16x32
constexpr int W_ENC1 = W_ENC0 + 16 * 32;  // 16x32
constexpr int W_ENCL = W_ENC1 + 16 * 32;  // 16x32
constexpr int W_HI0  = W_ENCL + 16 * 32;  // 16x32
constexpr int W_HI1  = W_HI0  + 16 * 32;  // 32x32
constexpr int W_HIR  = W_HI1  + 32 * 32;  // 64x32
constexpr int W_LO0  = W_HIR  + 64 * 32;  // 16x32
constexpr int W_LO1  = W_LO0  + 16 * 32;  // 32x32
constexpr int W_LOR  = W_LO1  + 32 * 32;  // 64x32
constexpr int W_TOT  = W_LOR  + 64 * 32;  // 8704 f16 = 17 KB

// f32 bias LDS offsets (padded to N tile multiples)
constexpr int B_ENC0 = 0;
constexpr int B_ENC1 = B_ENC0 + 16;
constexpr int B_ENCL = B_ENC1 + 16;
constexpr int B_HI0  = B_ENCL + 16;
constexpr int B_HI1  = B_HI0  + 16;
constexpr int B_HIR  = B_HI1  + 32;
constexpr int B_LO0  = B_HIR  + 64;
constexpr int B_LO1  = B_LO0  + 16;
constexpr int B_LOR  = B_LO1  + 32;
constexpr int B_TOT  = B_LOR  + 64;   // 272 floats

// Fill transposed, zero-padded weight matrix: dst[n*32+k] = (k<din && n<dout) ? w[k][n] : 0
__device__ inline void fill_w(_Float16* dst, const float* w, int din, int dout,
                              int npad, int tid) {
  const int total = npad * 32;
  for (int i = tid; i < total; i += NTHREADS) {
    const int n = i >> 5, k = i & 31;
    const float v = (n < dout && k < din) ? w[k * dout + n] : 0.0f;
    dst[i] = (_Float16)v;
  }
}

__device__ inline void fill_b(float* dst, const float* b, int dout, int npad, int tid) {
  for (int i = tid; i < npad; i += NTHREADS) dst[i] = (i < dout) ? b[i] : 0.0f;
}

__device__ inline v16h cat16(v8h lo, v8h hi) {
  v16h r;
#pragma unroll
  for (int i = 0; i < 8; ++i) { r[i] = lo[i]; r[i + 8] = hi[i]; }
  return r;
}

// A-matrix (16x32 f16): lanes 0-15 row=lane hold K0-7 / K16-23,
// lanes 16-31 row=lane-16 hold K8-15 / K24-31. act is row-major [16][32] f16.
__device__ inline v16h load_A(const _Float16* act, int lane) {
  const int r = lane & 15, h = lane >> 4;
  const _Float16* p = act + r * 32 + h * 8;
  return cat16(*(const v8h*)p, *(const v8h*)(p + 16));
}

// B-matrix (32x16 f16): lanes 0-15 col=lane hold K0-7 / K16-23; lanes 16-31 hold K8-15 / K24-31.
// wt is w^T stored [n][32] f16.
__device__ inline v16h load_Bm(const _Float16* wt, int nt, int lane) {
  const _Float16* p = wt + (nt * 16 + (lane & 15)) * 32 + (lane >> 4) * 8;
  return cat16(*(const v8h*)p, *(const v8h*)(p + 16));
}

__device__ inline v8f wmma16(v16h a, v16h b, v8f c) {
  return __builtin_amdgcn_wmma_f32_16x16x32_f16(
      /*neg_a=*/false, a, /*neg_b=*/false, b,
      /*c_mod=*/(short)0, c, /*reuse_a=*/false, /*reuse_b=*/false);
}

__device__ inline v8f bias_relu(v8f c, const float* bias, int nt, int lane) {
  const float b = bias[nt * 16 + (lane & 15)];
#pragma unroll
  for (int i = 0; i < 8; ++i) c[i] = fmaxf(c[i] + b, 0.0f);
  return c;
}

// C/D layout: VGPR i -> row = i + 8*(lane>=16), col = nt*16 + (lane&15)
__device__ inline void store_act(_Float16* act, v8f c, int nt, int lane) {
  const int h = lane >> 4, col = nt * 16 + (lane & 15);
#pragma unroll
  for (int i = 0; i < 8; ++i) act[(i + h * 8) * 32 + col] = (_Float16)c[i];
}

// One 16-wide layer (N<=16): act -> act (or zbuf), relu
__device__ inline void layer16(const _Float16* wt, const float* bias,
                               const _Float16* in_act, _Float16* out_act, int lane) {
  const v16h a = load_A(in_act, lane);
  v8f c = {};
  c = wmma16(a, load_Bm(wt, 0, lane), c);
  c = bias_relu(c, bias, 0, lane);
  store_act(out_act, c, 0, lane);
}

template <int NT>
__device__ inline void layerN(const _Float16* wt, const float* bias,
                              const _Float16* in_act, _Float16* out_act, int lane) {
  const v16h a = load_A(in_act, lane);  // loaded before any store (in==out safe)
#pragma unroll
  for (int t = 0; t < NT; ++t) {
    v8f c = {};
    c = wmma16(a, load_Bm(wt, t, lane), c);
    c = bias_relu(c, bias, t, lane);
    store_act(out_act, c, t, lane);
  }
}

// Final 32->64 layer: linear, store fp32 straight to global
__device__ inline void layer_out(const _Float16* wt, const float* bias,
                                 const _Float16* in_act, float* __restrict__ outp,
                                 int lane) {
  const v16h a = load_A(in_act, lane);
  const int h = lane >> 4, nl = lane & 15;
#pragma unroll
  for (int t = 0; t < 4; ++t) {
    v8f c = {};
    c = wmma16(a, load_Bm(wt, t, lane), c);
    const float b = bias[t * 16 + nl];
#pragma unroll
    for (int i = 0; i < 8; ++i)
      outp[(size_t)(i + h * 8) * kF + t * 16 + nl] = c[i] + b;
  }
}

__global__ __launch_bounds__(NTHREADS) void rancoders_wmma(
    const float* __restrict__ x, const int* __restrict__ rsamp,
    const float* __restrict__ ew0, const float* __restrict__ eb0,
    const float* __restrict__ ew1, const float* __restrict__ eb1,
    const float* __restrict__ ewl, const float* __restrict__ ebl,
    const float* __restrict__ hw0, const float* __restrict__ hb0,
    const float* __restrict__ hw1, const float* __restrict__ hb1,
    const float* __restrict__ hwr, const float* __restrict__ hbr,
    const float* __restrict__ lw0, const float* __restrict__ lb0,
    const float* __restrict__ lw1, const float* __restrict__ lb1,
    const float* __restrict__ lwr, const float* __restrict__ lbr,
    float* __restrict__ out) {
  __shared__ __align__(16) _Float16 sw[W_TOT];
  __shared__ __align__(16) float    sb[B_TOT];
  __shared__ __align__(16) int      sidx[kMF];
  __shared__ __align__(16) _Float16 sact[8][16 * 32];  // per-wave activation tile
  __shared__ __align__(16) _Float16 szl[8][16 * 32];   // per-wave latent tile

  const int tid  = threadIdx.x;
  const int e    = blockIdx.y;
  const int row0 = blockIdx.x * ROWS_PER_BLOCK;

  // ---- cooperative weight/bias staging (transpose + zero-pad to WMMA B layout)
  fill_w(sw + W_ENC0, ew0 + e * kMF * 8, kMF, 8, 16, tid);
  fill_w(sw + W_ENC1, ew1 + e * 8 * 4,   8,   4, 16, tid);
  fill_w(sw + W_ENCL, ewl + e * 4 * kL,  4,  kL, 16, tid);
  fill_w(sw + W_HI0,  hw0 + e * kL * 16, kL, 16, 16, tid);
  fill_w(sw + W_HI1,  hw1 + e * 16 * 32, 16, 32, 32, tid);
  fill_w(sw + W_HIR,  hwr + e * 32 * 64, 32, 64, 64, tid);
  fill_w(sw + W_LO0,  lw0 + e * kL * 16, kL, 16, 16, tid);
  fill_w(sw + W_LO1,  lw1 + e * 16 * 32, 16, 32, 32, tid);
  fill_w(sw + W_LOR,  lwr + e * 32 * 64, 32, 64, 64, tid);

  fill_b(sb + B_ENC0, eb0 + e * 8,  8,  16, tid);
  fill_b(sb + B_ENC1, eb1 + e * 4,  4,  16, tid);
  fill_b(sb + B_ENCL, ebl + e * kL, kL, 16, tid);
  fill_b(sb + B_HI0,  hb0 + e * 16, 16, 16, tid);
  fill_b(sb + B_HI1,  hb1 + e * 32, 32, 32, tid);
  fill_b(sb + B_HIR,  hbr + e * 64, 64, 64, tid);
  fill_b(sb + B_LO0,  lb0 + e * 16, 16, 16, tid);
  fill_b(sb + B_LO1,  lb1 + e * 32, 32, 32, tid);
  fill_b(sb + B_LOR,  lbr + e * 64, 64, 64, tid);

  if (tid < kMF) sidx[tid] = rsamp[e * kMF + tid];

  // Zero activation/latent buffers once (avoid NaN garbage in padded K lanes)
  for (int i = tid; i < 8 * 16 * 32 / 2; i += NTHREADS) {
    ((uint32_t*)sact)[i] = 0u;
    ((uint32_t*)szl)[i]  = 0u;
  }
  __syncthreads();

  const int wave = tid >> 5, lane = tid & 31;
  _Float16* act = sact[wave];
  _Float16* zb  = szl[wave];
  const int rtile = row0 + wave * 16;

  // ---- layer 0: gather x into A regs directly (MF=16 <= K half, rest zero)
  {
    const int r = lane & 15, h = lane >> 4;
    const float* xr = x + (size_t)(rtile + r) * kF;
    v16h a;
#pragma unroll
    for (int j = 0; j < 8; ++j) {
      a[j]     = (_Float16)xr[sidx[h * 8 + j]];  // K = h?8..15 : 0..7
      a[j + 8] = (_Float16)0.0f;                 // K = 16..31 pad
    }
    v8f c = {};
    c = wmma16(a, load_Bm(sw + W_ENC0, 0, lane), c);
    c = bias_relu(c, sb + B_ENC0, 0, lane);
    store_act(act, c, 0, lane);
  }

  // ---- encoder
  layer16(sw + W_ENC1, sb + B_ENC1, act, act, lane);
  layer16(sw + W_ENCL, sb + B_ENCL, act, zb,  lane);  // latent z

  float* outhi = out + ((size_t)e * kB + rtile) * kF;
  float* outlo = outhi + (size_t)kE * kB * kF;

  // ---- upper-bound decoder
  layer16  (sw + W_HI0, sb + B_HI0, zb,  act, lane);
  layerN<2>(sw + W_HI1, sb + B_HI1, act, act, lane);
  layer_out(sw + W_HIR, sb + B_HIR, act, outhi, lane);

  // ---- lower-bound decoder (restart from z)
  layer16  (sw + W_LO0, sb + B_LO0, zb,  act, lane);
  layerN<2>(sw + W_LO1, sb + B_LO1, act, act, lane);
  layer_out(sw + W_LOR, sb + B_LOR, act, outlo, lane);
}

}  // namespace

extern "C" void kernel_launch(void* const* d_in, const int* in_sizes, int n_in,
                              void* d_out, int out_size, void* d_ws, size_t ws_size,
                              hipStream_t stream) {
  (void)in_sizes; (void)n_in; (void)out_size; (void)d_ws; (void)ws_size;
  const float* x   = (const float*)d_in[0];
  const int*   rs  = (const int*)d_in[1];
  const float* ew0 = (const float*)d_in[2];
  const float* eb0 = (const float*)d_in[3];
  const float* ew1 = (const float*)d_in[4];
  const float* eb1 = (const float*)d_in[5];
  const float* ewl = (const float*)d_in[6];
  const float* ebl = (const float*)d_in[7];
  const float* hw0 = (const float*)d_in[8];
  const float* hb0 = (const float*)d_in[9];
  const float* hw1 = (const float*)d_in[10];
  const float* hb1 = (const float*)d_in[11];
  const float* hwr = (const float*)d_in[12];
  const float* hbr = (const float*)d_in[13];
  const float* lw0 = (const float*)d_in[14];
  const float* lb0 = (const float*)d_in[15];
  const float* lw1 = (const float*)d_in[16];
  const float* lb1 = (const float*)d_in[17];
  const float* lwr = (const float*)d_in[18];
  const float* lbr = (const float*)d_in[19];
  float* out = (float*)d_out;

  dim3 grid(kB / ROWS_PER_BLOCK, kE, 1);
  rancoders_wmma<<<grid, NTHREADS, 0, stream>>>(
      x, rs, ew0, eb0, ew1, eb1, ewl, ebl,
      hw0, hb0, hw1, hb1, hwr, hbr,
      lw0, lb0, lw1, lb1, lwr, lbr, out);
}
